// Head_7576322310288
// MI455X (gfx1250) — matrix operand order repaired and checked by reading, not compiled, not run on hardware
//
#include <hip/hip_runtime.h>
#include <hip/hip_bf16.h>
#include <math.h>

// Problem constants (reference: B,T,C,H = 8,2048,1024,64)
#define Bsz  8
#define Tseq 2048
#define Cdim 1024
#define Hdim 64
#define BT   (Bsz * Tseq)

typedef __attribute__((ext_vector_type(8)))  float        v8f;
typedef __attribute__((ext_vector_type(8)))  __bf16       v8bf;
typedef __attribute__((ext_vector_type(16))) __bf16       v16bf;
typedef __attribute__((ext_vector_type(4)))  unsigned int v4u;
typedef __attribute__((ext_vector_type(4)))  int          v4i;
typedef __attribute__((ext_vector_type(8)))  int          v8i;

static __device__ __forceinline__ v8f wmma_bf16(v16bf a, v16bf b, v8f c) {
    // D = A(16x32 bf16) * B(32x16 bf16) + C(16x16 f32)
    return __builtin_amdgcn_wmma_f32_16x16x32_bf16(
        /*neg_a=*/false, a, /*neg_b=*/false, b,
        /*c_mod=*/(short)0, c, /*reuse_a=*/false, /*reuse_b=*/false);
}

// Low 32 bits of a flat LDS pointer == byte offset in the workgroup's LDS
// allocation (ISA 10.2 aperture calc).
static __device__ __forceinline__ unsigned lds_off(const void* p) {
    return (unsigned)(uintptr_t)p;
}

// ---------------------------------------------------------------------------
// TDM: async 2D tile load Global -> LDS (bf16 elements, data_size code 1).
// D# packing per CDNA5 ISA ch.8:
//  group0: [1:0]=count=1, [63:32]=lds_addr, [120:64]=global_addr, [127:126]=2
//  group1: [17:16]=data_size, [79:48]=tensor_dim0, [111:80]=tensor_dim1,
//          [127:112]=tile_dim0, [143:128]=tile_dim1, [207:160]=dim0_stride
// tile_w elements are contiguous in memory and in LDS; tile_h rows strided
// by row_stride elements in memory, packed densely in LDS.
// ---------------------------------------------------------------------------
static __device__ __forceinline__ void tdm_load_2d_bf16(
    unsigned lds_addr, const void* gptr,
    unsigned tile_w, unsigned tile_h,
    unsigned tensor_w, unsigned tensor_h, unsigned row_stride)
{
    const unsigned long long ga = (unsigned long long)(uintptr_t)gptr;
    v4u g0;
    g0[0] = 1u;                                              // count=1 (valid)
    g0[1] = lds_addr;                                        // lds_addr
    g0[2] = (unsigned)(ga & 0xFFFFFFFFu);                    // global_addr lo
    g0[3] = (unsigned)((ga >> 32) & 0x1FFFFFFu) | (2u << 30);// addr[56:32]|type=2
    v8i g1;
    g1[0] = (int)(1u << 16);                                 // data_size=2B
    g1[1] = (int)((tensor_w & 0xFFFFu) << 16);               // tensor_dim0 lo16
    g1[2] = (int)((tensor_w >> 16) | ((tensor_h & 0xFFFFu) << 16));
    g1[3] = (int)((tensor_h >> 16) | (tile_w << 16));        // dim1 hi | tile_dim0
    g1[4] = (int)(tile_h & 0xFFFFu);                         // tile_dim1
    g1[5] = (int)row_stride;                                 // dim0_stride lo32
    g1[6] = 0;                                               // stride hi / dim1_stride
    g1[7] = 0;
    const v4i z4 = {0, 0, 0, 0};
#if __has_include(<hip/amd_detail/amd_gfx1250_TDM.h>)
    const v8i z8 = {0, 0, 0, 0, 0, 0, 0, 0};
    __builtin_amdgcn_tensor_load_to_lds(g0, g1, z4, z4, z8, 0);
#else
    __builtin_amdgcn_tensor_load_to_lds(g0, g1, z4, z4, 0);
#endif
}

// ---------------------------------------------------------------------------
// Kernel 1: Q = x@Wq, K = x@Wk, V = x@Wv   (M=BT, K=C, N=H), bf16 outputs.
// Q,K row-major [BT,H]; V transposed [B,H,T].
// grid = (BT/16, 3), block = 128 (4 waves); wave w owns n-tile w of H.
// wave32 fragment layouts used by this kernel (contiguous k, not ISA table order):
//   A 16x32 bf16: lane -> row M=lane&15; elem e -> K=hf*16+e
//   B 32x16 bf16: lane -> col N=lane&15; elem e -> K=hf*16+e
//   C/D 16x16 f32: VGPR r -> row M=r+hf*8, col N=lane&15
// ---------------------------------------------------------------------------
__global__ __launch_bounds__(128) void qkv_proj_kernel(
    const float* __restrict__ x,  const float* __restrict__ Wq,
    const float* __restrict__ Wk, const float* __restrict__ Wv,
    __bf16* __restrict__ Qb, __bf16* __restrict__ Kb, __bf16* __restrict__ Vt)
{
    const int lane = threadIdx.x & 31;
    const int wave = threadIdx.x >> 5;
    const int ln   = lane & 15;
    const int hf   = lane >> 4;
    const int m0   = blockIdx.x * 16;
    const int n0   = wave * 16;
    const int mat  = blockIdx.y;

    const float* W = (mat == 0) ? Wq : (mat == 1) ? Wk : Wv;
    const float* xrow = x + (size_t)(m0 + ln) * Cdim;

    v8f acc = {};
    for (int k0 = 0; k0 < Cdim; k0 += 32) {
        v16bf a, bw;
#pragma unroll
        for (int j = 0; j < 8; ++j) {
            a[j]     = (__bf16)xrow[k0 + hf * 16 + j];
            a[j + 8] = (__bf16)xrow[k0 + hf * 16 + 8 + j];
        }
        const float* wcol = W + (size_t)(k0 + hf * 16) * Hdim + (n0 + ln);
#pragma unroll
        for (int e = 0; e < 16; ++e)
            bw[e] = (__bf16)wcol[(size_t)e * Hdim];
        acc = wmma_bf16(a, bw, acc);
    }

#pragma unroll
    for (int r = 0; r < 8; ++r) {
        const int row = m0 + r + hf * 8;
        const int col = n0 + ln;
        const __bf16 val = (__bf16)acc[r];
        if (mat == 0) {
            Qb[(size_t)row * Hdim + col] = val;
        } else if (mat == 1) {
            Kb[(size_t)row * Hdim + col] = val;
        } else {
            const int b = row >> 11;
            const int t = row & (Tseq - 1);
            Vt[((size_t)b * Hdim + col) * Tseq + t] = val;
        }
    }
}

// ---------------------------------------------------------------------------
// Kernel 2: causal flash attention.
// grid = (T/64, B), block = 128 (4 waves). The block owns query rows
// [t0_blk, t0_blk+63]; wave w owns rows [t0_blk+16w, +16). Key tiles (32 keys)
// are streamed Global->LDS by the Tensor Data Mover (double-buffered, wave 0
// issues, TENSORcnt + barriers for handoff); all 4 waves consume the shared
// tile. The key loop is block-uniform; masking keeps causality per wave.
// ---------------------------------------------------------------------------
__global__ __launch_bounds__(128) void attn_kernel(
    const __bf16* __restrict__ Qb, const __bf16* __restrict__ Kb,
    const __bf16* __restrict__ Vt, float* __restrict__ out)
{
    __shared__ __align__(32) __bf16 Kt[2][32][64];  // [buf][key][h]     4KB ea
    __shared__ __align__(32) __bf16 Vs[2][64][32];  // [buf][h][key]     4KB ea
    __shared__ __align__(32) __bf16 Pbuf[4][16][32];

    const int lane   = threadIdx.x & 31;
    const int wave   = threadIdx.x >> 5;
    const int ln     = lane & 15;
    const int hf     = lane >> 4;
    const int b      = blockIdx.y;
    const int t0blk  = blockIdx.x * 64;
    const int t0     = t0blk + wave * 16;
    const int send   = t0blk + 63;          // last key column the block needs
    const float scale = 0.125f;             // 1/sqrt(64)

    // ---- Q A-fragments for both 32-wide h-chunks ----
    const __bf16* qrow = Qb + ((size_t)(b * Tseq + t0 + ln)) * Hdim;
    v16bf aq0, aq1;
#pragma unroll
    for (int j = 0; j < 8; ++j) {
        aq0[j]     = qrow[hf * 16 + j];
        aq0[j + 8] = qrow[hf * 16 + 8 + j];
        aq1[j]     = qrow[32 + hf * 16 + j];
        aq1[j + 8] = qrow[40 + hf * 16 + j];
    }

    float m_s[8], l_s[8], alpha[8];
    v8f o[4] = {{}, {}, {}, {}};
#pragma unroll
    for (int r = 0; r < 8; ++r) { m_s[r] = -__builtin_inff(); l_s[r] = 0.0f; }

    const __bf16* kbase = Kb + (size_t)b * Tseq * Hdim;   // [T,H]
    const __bf16* vbase = Vt + (size_t)b * Hdim * Tseq;   // [H,T]

    // Prime the pipeline: tile 0 into buffer 0 (TDM ignores EXEC; wave 0 only).
    if (wave == 0) {
        tdm_load_2d_bf16(lds_off(&Kt[0][0][0]), kbase, 64, 32, Hdim, Tseq, Hdim);
        tdm_load_2d_bf16(lds_off(&Vs[0][0][0]), vbase, 32, 64, Tseq, Hdim, Tseq);
    }

    for (int s0 = 0; s0 <= send; s0 += 32) {
        const int buf = (s0 >> 5) & 1;
        if (wave == 0) __builtin_amdgcn_s_wait_tensorcnt(0);
        __syncthreads();                               // tile `buf` visible

        if (wave == 0 && s0 + 32 <= send) {            // prefetch next tile
            const int nb = buf ^ 1;
            tdm_load_2d_bf16(lds_off(&Kt[nb][0][0]),
                             kbase + (size_t)(s0 + 32) * Hdim,
                             64, 32, Hdim, Tseq, Hdim);
            tdm_load_2d_bf16(lds_off(&Vs[nb][0][0]),
                             vbase + (s0 + 32),
                             32, 64, Tseq, Hdim, Tseq);
        }

        // Wave-uniform predicate: whole wave active or idle -> EXEC all-ones
        // around every WMMA.
        if (s0 <= t0 + 15) {
            // ---- S = Q K^T (16x32): B-fragments straight from LDS ----
            v8f sa = {}, sb = {};
            {
                const v16bf bk00 = *(const v16bf*)&Kt[buf][ln][hf * 16];
                const v16bf bk01 = *(const v16bf*)&Kt[buf][16 + ln][hf * 16];
                const v16bf bk10 = *(const v16bf*)&Kt[buf][ln][32 + hf * 16];
                const v16bf bk11 = *(const v16bf*)&Kt[buf][16 + ln][32 + hf * 16];
                sa = wmma_bf16(aq0, bk00, sa);
                sb = wmma_bf16(aq0, bk01, sb);
                sa = wmma_bf16(aq1, bk10, sa);
                sb = wmma_bf16(aq1, bk11, sb);
            }

            // ---- causal mask + online softmax ----
            const int c0 = s0 + ln;
            const int c1 = s0 + 16 + ln;
#pragma unroll
            for (int r = 0; r < 8; ++r) {
                const int trow = t0 + r + hf * 8;
                float v0 = (c0 <= trow) ? sa[r] * scale : -__builtin_inff();
                float v1 = (c1 <= trow) ? sb[r] * scale : -__builtin_inff();
                float mx = fmaxf(v0, v1);
#pragma unroll
                for (int m = 1; m < 16; m <<= 1)
                    mx = fmaxf(mx, __shfl_xor(mx, m, 32));
                const float mnew = fmaxf(m_s[r], mx);
                const float al = __expf(m_s[r] - mnew);
                const float p0 = __expf(v0 - mnew);
                const float p1 = __expf(v1 - mnew);
                float sum = p0 + p1;
#pragma unroll
                for (int m = 1; m < 16; m <<= 1)
                    sum += __shfl_xor(sum, m, 32);
                l_s[r] = l_s[r] * al + sum;
                m_s[r] = mnew;
                alpha[r] = al;
                Pbuf[wave][r + hf * 8][ln]      = (__bf16)p0;
                Pbuf[wave][r + hf * 8][16 + ln] = (__bf16)p1;
            }
#pragma unroll
            for (int r = 0; r < 8; ++r) {
                o[0][r] *= alpha[r]; o[1][r] *= alpha[r];
                o[2][r] *= alpha[r]; o[3][r] *= alpha[r];
            }

            // Same-wave DS ops complete in order; fence the scheduler.
            __builtin_amdgcn_wave_barrier();

            // ---- P (C-layout in LDS) -> A-fragment ----
            const v8bf plo = *(const v8bf*)&Pbuf[wave][ln][hf * 16];
            const v8bf phi = *(const v8bf*)&Pbuf[wave][ln][hf * 16 + 8];
            v16bf ap;
#pragma unroll
            for (int j = 0; j < 8; ++j) { ap[j] = plo[j]; ap[j + 8] = phi[j]; }

            // ---- O += P V : V B-fragments from LDS ----
#pragma unroll
            for (int j = 0; j < 4; ++j) {
                const v16bf bv = *(const v16bf*)&Vs[buf][j * 16 + ln][hf * 16];
                o[j] = wmma_bf16(ap, bv, o[j]);
            }
        }

        __syncthreads();   // all reads of `buf` done before it is re-filled
    }

    // ---- normalize and write fp32 output [B,T,H] ----
#pragma unroll
    for (int r = 0; r < 8; ++r) {
        const float inv = 1.0f / l_s[r];
        const size_t row = (size_t)b * Tseq + t0 + r + hf * 8;
#pragma unroll
        for (int j = 0; j < 4; ++j)
            out[row * Hdim + j * 16 + ln] = o[j][r] * inv;
    }
}

// ---------------------------------------------------------------------------
extern "C" void kernel_launch(void* const* d_in, const int* in_sizes, int n_in,
                              void* d_out, int out_size, void* d_ws, size_t ws_size,
                              hipStream_t stream) {
    const float* x  = (const float*)d_in[0];
    const float* Wq = (const float*)d_in[1];
    const float* Wk = (const float*)d_in[2];
    const float* Wv = (const float*)d_in[3];
    float* out = (float*)d_out;

    // Workspace: bf16 Q [BT,H], K [BT,H], V^T [B,H,T]  (3 * 2 MB = 6 MB)
    __bf16* Qb = (__bf16*)d_ws;
    __bf16* Kb = Qb + (size_t)BT * Hdim;
    __bf16* Vt = Kb + (size_t)BT * Hdim;

    dim3 g1(BT / 16, 3), b1(128);
    qkv_proj_kernel<<<g1, b1, 0, stream>>>(x, Wq, Wk, Wv, Qb, Kb, Vt);

    dim3 g2(Tseq / 64, Bsz), b2(128);
    attn_kernel<<<g2, b2, 0, stream>>>(Qb, Kb, Vt, out);
}
